// PolicyNet_64381559767613
// MI455X (gfx1250) — compile-verified, matrix operand order checked by reading, not run on hardware
//
#include <hip/hip_runtime.h>
#include <cstddef>
#include <cstdint>

// ---------------------------------------------------------------------------
// Problem constants (PolicyNet): M actions, N question tokens, D embed dim
// ---------------------------------------------------------------------------
#define M_ACT 16384
#define N_TOK 128
#define DIM   2048
#define KT_D  (DIM / 32)    // 64 k-tiles over the embedding dim
#define KT_N  (N_TOK / 32)  // 4 k-tiles over the token dim
#define JT_D  (DIM / 16)    // 128 col-tiles over the embedding dim
#define NT_N  (N_TOK / 16)  // 8 col-tiles over tokens
#define FRAG_ELEMS 512      // 32 lanes * 16 bf16 per WMMA operand tile

typedef __bf16 bf16_t;
typedef __attribute__((ext_vector_type(16))) __bf16 v16bf;
typedef __attribute__((ext_vector_type(8)))  float  v8f;

// ---------------------------------------------------------------------------
// WMMA helpers (CDNA5 wave32, V_WMMA_F32_16X16X32_BF16, f32 accumulate)
// VGPR layouts per CDNA5 ISA 7.12.2:
//   A (16x32 bf16): lane L holds row m=L&15; K packed 2/VGPR, hi-half lanes +8
//   B (32x16 bf16): lane L holds col n=L&15; K packed like A
//   C/D (16x16 f32): lane L holds col n=L&15, rows (L>>4)*8 + r  (r = VGPR idx)
// ---------------------------------------------------------------------------
__device__ __forceinline__ v8f wmma_bf16(v16bf a, v16bf b, v8f c) {
  return __builtin_amdgcn_wmma_f32_16x16x32_bf16(
      /*neg_a=*/false, a, /*neg_b=*/false, b,
      /*c_mod=*/(short)0, c, /*reuse_a=*/false, /*reuse_b=*/false);
}

// Map (lane, element e) of a 16x32 A-frag or 32x16 B-frag to (row/col, kk)
__device__ __forceinline__ void frag_coords(int lane, int e, int& mn, int& kk) {
  const int hk = (lane >> 4) * 8;
  const int v  = e >> 1;
  kk = ((v < 4) ? (2 * v) : (16 + 2 * (v - 4))) + hk + (e & 1);
  mn = lane & 15;
}

// Fragment-packed operand: one aligned 32-byte vector load per lane.
__device__ __forceinline__ v16bf load_frag_packed(const bf16_t* base,
                                                  size_t fragIdx, int lane) {
  return *reinterpret_cast<const v16bf*>(base +
                                         (fragIdx * 32 + (size_t)lane) * 16);
}

// A-fragment gathered from LDS (row-major bf16, stride ld) — pairs merge to
// ds_load_b32 / ds_load_2addr_b32.
__device__ __forceinline__ v16bf load_frag_a_lds(const bf16_t* src, int ld,
                                                 int lane) {
  const int m  = lane & 15;
  const int hk = (lane >> 4) * 8;
  v16bf a;
#pragma unroll
  for (int v = 0; v < 8; ++v) {
    const int kk = ((v < 4) ? (2 * v) : (16 + 2 * (v - 4))) + hk;
    a[2 * v]     = src[m * ld + kk];
    a[2 * v + 1] = src[m * ld + kk + 1];
  }
  return a;
}

__device__ __forceinline__ void store_frag_f32(float* dst, int ld, int lane,
                                               v8f acc) {
  const int n  = lane & 15;
  const int mb = (lane >> 4) * 8;
#pragma unroll
  for (int r = 0; r < 8; ++r) dst[(mb + r) * ld + n] = acc[r];
}

__device__ __forceinline__ void store_frag_bf16(bf16_t* dst, int ld, int lane,
                                                v8f acc) {
  const int n  = lane & 15;
  const int mb = (lane >> 4) * 8;
#pragma unroll
  for (int r = 0; r < 8; ++r) dst[(mb + r) * ld + n] = (bf16_t)acc[r];
}

// ---------------------------------------------------------------------------
// Prep kernels: convert f32 sources into WMMA-fragment-packed bf16 layouts.
// Output layout: [tile][lane(32)][elem(16)] -> lane's 16 bf16 are contiguous.
// ---------------------------------------------------------------------------

// Actions as A-fragments: tile = mtile*KT_D + ktile
__global__ void k_pack_a_act(const float* __restrict__ act,
                             bf16_t* __restrict__ dst) {
  const size_t total = (size_t)(M_ACT / 16) * KT_D * FRAG_ELEMS;
  size_t i = (size_t)blockIdx.x * blockDim.x + threadIdx.x;
  const size_t stride = (size_t)gridDim.x * blockDim.x;
  for (; i < total; i += stride) {
    const int e    = (int)(i & 15);
    const int lane = (int)((i >> 4) & 31);
    const size_t tile = i >> 9;
    const int kt = (int)(tile % KT_D);
    const int mt = (int)(tile / KT_D);
    int m, kk;
    frag_coords(lane, e, m, kk);
    dst[i] = (bf16_t)act[(size_t)(mt * 16 + m) * DIM + kt * 32 + kk];
  }
}

// Q^T as B-fragments for scores: B[k,n] = Q[n,k]; tile = ntile*KT_D + ktile
__global__ void k_pack_b_qt(const float* __restrict__ q,
                            bf16_t* __restrict__ dst) {
  const size_t total = (size_t)NT_N * KT_D * FRAG_ELEMS;
  size_t i = (size_t)blockIdx.x * blockDim.x + threadIdx.x;
  const size_t stride = (size_t)gridDim.x * blockDim.x;
  for (; i < total; i += stride) {
    const int e    = (int)(i & 15);
    const int lane = (int)((i >> 4) & 31);
    const size_t tile = i >> 9;
    const int kt = (int)(tile % KT_D);
    const int nt = (int)(tile / KT_D);
    int n, kk;
    frag_coords(lane, e, n, kk);
    dst[i] = (bf16_t)q[(size_t)(nt * 16 + n) * DIM + kt * 32 + kk];
  }
}

// Q as B-fragments for attn@Q: B[k,j] = Q[k,j]; tile = jtile*KT_N + ktile
__global__ void k_pack_b_q(const float* __restrict__ q,
                           bf16_t* __restrict__ dst) {
  const size_t total = (size_t)JT_D * KT_N * FRAG_ELEMS;
  size_t i = (size_t)blockIdx.x * blockDim.x + threadIdx.x;
  const size_t stride = (size_t)gridDim.x * blockDim.x;
  for (; i < total; i += stride) {
    const int e    = (int)(i & 15);
    const int lane = (int)((i >> 4) & 31);
    const size_t tile = i >> 9;
    const int kt = (int)(tile % KT_N);
    const int jt = (int)(tile / KT_N);
    int n, kk;
    frag_coords(lane, e, n, kk);
    dst[i] = (bf16_t)q[(size_t)(kt * 32 + kk) * DIM + jt * 16 + n];
  }
}

// W1 "question half" as B-fragments: B[k,j] = W1[j, D+k]; tile = jt*KT_D + kt
__global__ void k_pack_b_w1(const float* __restrict__ W1,
                            bf16_t* __restrict__ dst) {
  const size_t total = (size_t)JT_D * KT_D * FRAG_ELEMS;
  size_t i = (size_t)blockIdx.x * blockDim.x + threadIdx.x;
  const size_t stride = (size_t)gridDim.x * blockDim.x;
  for (; i < total; i += stride) {
    const int e    = (int)(i & 15);
    const int lane = (int)((i >> 4) & 31);
    const size_t tile = i >> 9;
    const int kt = (int)(tile % KT_D);
    const int jt = (int)(tile / KT_D);
    int n, kk;
    frag_coords(lane, e, n, kk);
    dst[i] = (bf16_t)W1[(size_t)(jt * 16 + n) * (2 * DIM) + DIM + kt * 32 + kk];
  }
}

// W2 as B-fragments: B[k,j] = W2[j,k]; tile = jt*KT_D + kt
__global__ void k_pack_b_w2(const float* __restrict__ W2,
                            bf16_t* __restrict__ dst) {
  const size_t total = (size_t)JT_D * KT_D * FRAG_ELEMS;
  size_t i = (size_t)blockIdx.x * blockDim.x + threadIdx.x;
  const size_t stride = (size_t)gridDim.x * blockDim.x;
  for (; i < total; i += stride) {
    const int e    = (int)(i & 15);
    const int lane = (int)((i >> 4) & 31);
    const size_t tile = i >> 9;
    const int kt = (int)(tile % KT_D);
    const int jt = (int)(tile / KT_D);
    int n, kk;
    frag_coords(lane, e, n, kk);
    dst[i] = (bf16_t)W2[(size_t)(jt * 16 + n) * DIM + kt * 32 + kk];
  }
}

// b1'[j] = b1[j] + dot(W1[j, 0:D], hist)  (history half of W1 folded to bias)
__global__ void k_fold_bias(const float* __restrict__ W1,
                            const float* __restrict__ b1,
                            const float* __restrict__ hist,
                            float* __restrict__ b1p) {
  __shared__ float buf[256];
  const int j = blockIdx.x;
  const int tid = threadIdx.x;
  const float* row = W1 + (size_t)j * (2 * DIM);
  float acc = 0.f;
  for (int k = tid; k < DIM; k += 256) acc += row[k] * hist[k];
  buf[tid] = acc;
  __syncthreads();
  for (int o = 128; o >= 1; o >>= 1) {
    if (tid < o) buf[tid] += buf[tid + o];
    __syncthreads();
  }
  if (tid == 0) b1p[j] = b1[j] + buf[0];
}

// ---------------------------------------------------------------------------
// Fused main kernel: one 16-row action tile per workgroup (8 wave32 waves).
// LDS (dynamic): scores f32 [16,128] | attn bf16 [16,128] |
//                q bf16 [16,2048] | h bf16 [16,2048] | srow f32 [16]
// ---------------------------------------------------------------------------
#define SMEM_SCORES 0
#define SMEM_ATTN   (16 * 128 * 4)
#define SMEM_Q      (SMEM_ATTN + 16 * 128 * 2)
#define SMEM_H      (SMEM_Q + 16 * 2048 * 2)
#define SMEM_SROW   (SMEM_H + 16 * 2048 * 2)
#define SMEM_TOTAL  (SMEM_SROW + 16 * 4)

__global__ __launch_bounds__(256) void k_policy_main(
    const bf16_t* __restrict__ A16P,  // A-frag packed actions
    const bf16_t* __restrict__ QP,    // B-frag packed Q (attn@Q)
    const bf16_t* __restrict__ QTP,   // B-frag packed Q^T (scores)
    const bf16_t* __restrict__ W1P,   // B-frag packed W1 question-half
    const bf16_t* __restrict__ W2P,   // B-frag packed W2
    const float*  __restrict__ b1p,   // [D]
    const float*  __restrict__ b2,    // [D]
    const float*  __restrict__ actF,  // [M, D] f32 actions (final dot)
    float* __restrict__ sOut)         // [M] semantic scores
{
  extern __shared__ char smem[];
  float*  sS    = (float*)(smem + SMEM_SCORES);
  bf16_t* sAttn = (bf16_t*)(smem + SMEM_ATTN);
  bf16_t* sQ    = (bf16_t*)(smem + SMEM_Q);
  bf16_t* sH    = (bf16_t*)(smem + SMEM_H);
  float*  sRow  = (float*)(smem + SMEM_SROW);

  const int tid  = threadIdx.x;
  const int wave = tid >> 5;
  const int lane = tid & 31;
  const int r0   = blockIdx.x * 16;

  if (tid < 16) sRow[tid] = 0.f;

  // ---- Stage A: scores[16,128] = act_tile @ Q^T  (each wave: one 16-col tile)
  {
    const size_t aBase = (size_t)blockIdx.x * KT_D;  // A frag index base
    const size_t bBase = (size_t)wave * KT_D;        // B frag index base
    v8f acc = {};
    for (int kt = 0; kt < KT_D; ++kt) {
      v16bf a = load_frag_packed(A16P, aBase + kt, lane);
      v16bf b = load_frag_packed(QTP, bBase + kt, lane);
      acc = wmma_bf16(a, b, acc);
    }
    store_frag_f32(sS + wave * 16, 128, lane, acc);
  }
  __syncthreads();

  // ---- Row softmax over 128 tokens (16 threads per row, wave16 shuffles)
  {
    const int row = tid >> 4;
    const int col = tid & 15;
    const float* Sr = sS + row * 128;
    float e[8];
    float mx = -3.0e38f;
#pragma unroll
    for (int j = 0; j < 8; ++j) {
      e[j] = Sr[col + 16 * j];
      mx = fmaxf(mx, e[j]);
    }
#pragma unroll
    for (int m = 8; m >= 1; m >>= 1) mx = fmaxf(mx, __shfl_xor(mx, m, 16));
    float sum = 0.f;
#pragma unroll
    for (int j = 0; j < 8; ++j) {
      e[j] = __expf(e[j] - mx);
      sum += e[j];
    }
#pragma unroll
    for (int m = 8; m >= 1; m >>= 1) sum += __shfl_xor(sum, m, 16);
    const float inv = 1.f / sum;
#pragma unroll
    for (int j = 0; j < 8; ++j)
      sAttn[row * 128 + col + 16 * j] = (bf16_t)(e[j] * inv);
  }
  __syncthreads();

  // ---- Stage A2: q[16,2048] = attn[16,128] @ Q[128,2048]
  for (int t = wave; t < JT_D; t += 8) {
    const size_t bBase = (size_t)t * KT_N;
    v8f acc = {};
#pragma unroll
    for (int kt = 0; kt < KT_N; ++kt) {
      v16bf a = load_frag_a_lds(sAttn + kt * 32, 128, lane);
      v16bf b = load_frag_packed(QP, bBase + kt, lane);
      acc = wmma_bf16(a, b, acc);
    }
    store_frag_bf16(sQ + t * 16, DIM, lane, acc);
  }
  __syncthreads();

  // ---- Stage B: h = relu(q @ W1hat^T + b1')  (W1P streamed linearly from L2)
  for (int t = wave; t < JT_D; t += 8) {
    const size_t bBase = (size_t)t * KT_D;
    v8f acc = {};
    for (int kt = 0; kt < KT_D; ++kt) {
      if (kt + 1 < KT_D)
        __builtin_prefetch(W1P + (bBase + kt + 1) * FRAG_ELEMS, 0, 0);
      v16bf a = load_frag_a_lds(sQ + kt * 32, DIM, lane);
      v16bf b = load_frag_packed(W1P, bBase + kt, lane);
      acc = wmma_bf16(a, b, acc);
    }
    const int n  = lane & 15;
    const int mb = (lane >> 4) * 8;
    const float bias = b1p[t * 16 + n];
#pragma unroll
    for (int r = 0; r < 8; ++r) {
      float v = fmaxf(acc[r] + bias, 0.f);
      sH[(mb + r) * DIM + t * 16 + n] = (bf16_t)v;
    }
  }
  __syncthreads();

  // ---- Stage C: qwh = h @ W2^T + b2, fused with row-dot against actions
  {
    float sp[8] = {0.f, 0.f, 0.f, 0.f, 0.f, 0.f, 0.f, 0.f};
    const int n  = lane & 15;
    const int mb = (lane >> 4) * 8;
    for (int t = wave; t < JT_D; t += 8) {
      const size_t bBase = (size_t)t * KT_D;
      v8f acc = {};
      for (int kt = 0; kt < KT_D; ++kt) {
        if (kt + 1 < KT_D)
          __builtin_prefetch(W2P + (bBase + kt + 1) * FRAG_ELEMS, 0, 0);
        v16bf a = load_frag_a_lds(sH + kt * 32, DIM, lane);
        v16bf b = load_frag_packed(W2P, bBase + kt, lane);
        acc = wmma_bf16(a, b, acc);
      }
      const float bias = b2[t * 16 + n];
#pragma unroll
      for (int r = 0; r < 8; ++r) {
        const float v = acc[r] + bias;
        sp[r] += v * actF[(size_t)(r0 + mb + r) * DIM + t * 16 + n];
      }
    }
    // reduce the 16 column-lanes holding the same rows
#pragma unroll
    for (int r = 0; r < 8; ++r) {
#pragma unroll
      for (int m = 8; m >= 1; m >>= 1) sp[r] += __shfl_xor(sp[r], m, 16);
    }
    if ((lane & 15) == 0) {
#pragma unroll
      for (int r = 0; r < 8; ++r) atomicAdd(&sRow[mb + r], sp[r]);
    }
  }
  __syncthreads();

  if (tid < 16) sOut[r0 + tid] = sRow[tid];
}

// ---------------------------------------------------------------------------
// Global softmax over M actions
// ---------------------------------------------------------------------------
__global__ void k_soft_reduce(const float* __restrict__ s,
                              float* __restrict__ red) {
  __shared__ float buf[256];
  const int tid = threadIdx.x;
  float mx = -3.0e38f;
  for (int i = tid; i < M_ACT; i += 256) mx = fmaxf(mx, s[i]);
  buf[tid] = mx;
  __syncthreads();
  for (int o = 128; o >= 1; o >>= 1) {
    if (tid < o) buf[tid] = fmaxf(buf[tid], buf[tid + o]);
    __syncthreads();
  }
  mx = buf[0];
  __syncthreads();
  float sum = 0.f;
  for (int i = tid; i < M_ACT; i += 256) sum += __expf(s[i] - mx);
  buf[tid] = sum;
  __syncthreads();
  for (int o = 128; o >= 1; o >>= 1) {
    if (tid < o) buf[tid] += buf[tid + o];
    __syncthreads();
  }
  if (tid == 0) {
    red[0] = mx;
    red[1] = buf[0];
  }
}

__global__ void k_soft_norm(const float* __restrict__ s,
                            const float* __restrict__ red,
                            float* __restrict__ out) {
  const int i = blockIdx.x * blockDim.x + threadIdx.x;
  if (i < M_ACT) out[i] = __expf(s[i] - red[0]) / red[1];
}

// ---------------------------------------------------------------------------
// Host-side launch (all on `stream`, graph-capture safe)
// ---------------------------------------------------------------------------
extern "C" void kernel_launch(void* const* d_in, const int* in_sizes, int n_in,
                              void* d_out, int out_size, void* d_ws,
                              size_t ws_size, hipStream_t stream) {
  (void)in_sizes; (void)n_in; (void)out_size; (void)ws_size;
  const float* act  = (const float*)d_in[0];  // [M, D]
  const float* qT   = (const float*)d_in[1];  // [N, D]
  const float* hist = (const float*)d_in[2];  // [D]
  const float* W1   = (const float*)d_in[3];  // [D, 2D]
  const float* b1   = (const float*)d_in[4];  // [D]
  const float* W2   = (const float*)d_in[5];  // [D, D]
  const float* b2   = (const float*)d_in[6];  // [D]
  float* out = (float*)d_out;

  // Workspace carve-up (bytes), 256-aligned
  char* ws = (char*)d_ws;
  size_t off = 0;
  auto carve = [&](size_t bytes) {
    char* p = ws + off;
    off += (bytes + 255) & ~(size_t)255;
    return p;
  };
  bf16_t* A16P = (bf16_t*)carve((size_t)M_ACT * DIM * 2);          // 64 MiB
  bf16_t* QP   = (bf16_t*)carve((size_t)JT_D * KT_N * FRAG_ELEMS * 2);
  bf16_t* QTP  = (bf16_t*)carve((size_t)NT_N * KT_D * FRAG_ELEMS * 2);
  bf16_t* W1P  = (bf16_t*)carve((size_t)JT_D * KT_D * FRAG_ELEMS * 2);  // 8 MiB
  bf16_t* W2P  = (bf16_t*)carve((size_t)JT_D * KT_D * FRAG_ELEMS * 2);  // 8 MiB
  float*  B1P  = (float*)carve((size_t)DIM * 4);
  float*  SV   = (float*)carve((size_t)M_ACT * 4);
  float*  RED  = (float*)carve(2 * 4);

  // Prep: fragment-pack all WMMA operands (one-time, coalesced writes)
  k_pack_a_act<<<8192, 256, 0, stream>>>(act, A16P);
  k_pack_b_qt<<<1024, 256, 0, stream>>>(qT, QTP);
  k_pack_b_q<<<1024, 256, 0, stream>>>(qT, QP);
  k_pack_b_w1<<<8192, 256, 0, stream>>>(W1, W1P);
  k_pack_b_w2<<<8192, 256, 0, stream>>>(W2, W2P);
  k_fold_bias<<<DIM, 256, 0, stream>>>(W1, b1, hist, B1P);

  // Fused attention + MLP + action dot (WMMA), 16 rows per workgroup
  k_policy_main<<<M_ACT / 16, 256, SMEM_TOTAL, stream>>>(
      A16P, QP, QTP, W1P, W2P, B1P, b2, act, SV);

  // Softmax over actions
  k_soft_reduce<<<1, 256, 0, stream>>>(SV, RED);
  k_soft_norm<<<M_ACT / 256, 256, 0, stream>>>(SV, RED, out);
}
